// Guide_32366873543262
// MI455X (gfx1250) — compile-verified, matrix operand order checked by reading, not run on hardware
//
#include <hip/hip_runtime.h>
#include <hip/hip_bf16.h>
#include <math.h>

// ---- CDNA5 WMMA types ----
typedef __attribute__((ext_vector_type(16))) __bf16 v16bf;
typedef __attribute__((ext_vector_type(8)))  __bf16 v8bf;
typedef __attribute__((ext_vector_type(4)))  __bf16 v4bf;
typedef __attribute__((ext_vector_type(8)))  float  v8f;

#define B_    2048
#define C_    50
#define V_    512
#define E_    128
#define MS_   4
#define KATT_ 7   // 2*MS-1

// gfx1250 async global->LDS path (ASYNCcnt), guarded so the file compiles on
// any toolchain; fallback is a plain synchronous copy.
#if defined(__AMDGCN__) && __has_builtin(__builtin_amdgcn_global_load_async_to_lds_b128) && __has_builtin(__builtin_amdgcn_s_wait_asynccnt)
#define USE_ASYNC_LDS 1
#else
#define USE_ASYNC_LDS 0
#endif

#if USE_ASYNC_LDS
// Exact param types per hipcc diagnostic: pointee is a GCC-style int4 vector;
// src is addrspace(1) ("__device__"), dst is addrspace(3) ("__shared__").
typedef int v4i_gcc __attribute__((vector_size(4 * sizeof(int))));
typedef __attribute__((address_space(1))) v4i_gcc* as1_v4i;
typedef __attribute__((address_space(3))) v4i_gcc* as3_v4i;
#define ASYNC_CP16(gsrc, ldst) \
  __builtin_amdgcn_global_load_async_to_lds_b128((as1_v4i)(gsrc), (as3_v4i)(ldst), 0, 0)
#endif

static __device__ __forceinline__ v16bf concat8(v8bf lo, v8bf hi) {
  return __builtin_shufflevector(lo, hi, 0,1,2,3,4,5,6,7,8,9,10,11,12,13,14,15);
}

static __device__ __forceinline__ v4bf cvt4(float4 v) {
  v4bf r = { (__bf16)v.x, (__bf16)v.y, (__bf16)v.z, (__bf16)v.w };
  return r;
}

// ---------------------------------------------------------------------------
// Kernel 1: feat_emb[b,c,e] = message[b,c,:] @ feat_w[c,:,e] + feat_b[c,e]
// 50 GEMMs of [2048x512]x[512x128], bf16 WMMA, f32 accumulate.
// grid: (B/128, C), block: 256 (8 waves). Wave w -> rows w*16..w*16+15, all 8 N-tiles.
// ---------------------------------------------------------------------------
__global__ __launch_bounds__(256)
void feat_gemm_kernel(const float* __restrict__ msg, const float* __restrict__ fw,
                      const float* __restrict__ fb, float* __restrict__ femb) {
  __shared__ __bf16 As[128 * 32];   // [m][k]
  __shared__ __bf16 Bs[128 * 32];   // [n][k] (transposed on stage-in)
  const int m0   = blockIdx.x * 128;
  const int ch   = blockIdx.y;
  const int tid  = threadIdx.x;
  const int wid  = tid >> 5;
  const int lane = tid & 31;
  const int lhalf = lane >> 4;     // 0 / 1
  const int l15   = lane & 15;

  v8f acc[8];
  v8f zero = {};
#pragma unroll
  for (int i = 0; i < 8; ++i) acc[i] = zero;

  for (int kt = 0; kt < V_ / 32; ++kt) {
    const int k0 = kt * 32;
    // stage A (message rows, fp32 -> bf16, packed 64-bit DS stores)
#pragma unroll
    for (int q = 0; q < 4; ++q) {
      int idx = tid * 16 + q * 4;
      int row = idx >> 5, col = idx & 31;
      const float* src = &msg[((size_t)(m0 + row) * C_ + ch) * V_ + k0 + col];
      const float4 v = *(const float4*)src;
      *(v4bf*)&As[row * 32 + col] = cvt4(v);
      if (kt + 1 < V_ / 32) __builtin_prefetch(src + 32, 0, 1);  // next K-chunk
    }
    // stage B transposed (feat_w[ch][k][n] -> Bs[n][k])
#pragma unroll
    for (int q = 0; q < 4; ++q) {
      int idx = tid * 16 + q * 4;
      int k = idx >> 7, n = idx & 127;
      const float4 v = *(const float4*)&fw[((size_t)ch * V_ + k0 + k) * E_ + n];
      Bs[(n + 0) * 32 + k] = (__bf16)v.x;
      Bs[(n + 1) * 32 + k] = (__bf16)v.y;
      Bs[(n + 2) * 32 + k] = (__bf16)v.z;
      Bs[(n + 3) * 32 + k] = (__bf16)v.w;
    }
    __syncthreads();

    const int kb = lhalf * 8;                 // low lanes: K 0-7/16-23, high: 8-15/24-31
    const int am = wid * 16 + l15;
    v16bf a = concat8(*(const v8bf*)&As[am * 32 + kb],
                      *(const v8bf*)&As[am * 32 + kb + 16]);
#pragma unroll
    for (int nt = 0; nt < 8; ++nt) {
      int n = nt * 16 + l15;
      v16bf b = concat8(*(const v8bf*)&Bs[n * 32 + kb],
                        *(const v8bf*)&Bs[n * 32 + kb + 16]);
      acc[nt] = __builtin_amdgcn_wmma_f32_16x16x32_bf16(
          false, a, false, b, (short)0, acc[nt], false, false);
    }
    __syncthreads();
  }

  // epilogue: D layout -> VGPR r: lanes0-15 M=r, lanes16-31 M=r+8
#pragma unroll
  for (int nt = 0; nt < 8; ++nt) {
    int col = nt * 16 + l15;
    float bias = fb[ch * E_ + col];
#pragma unroll
    for (int r = 0; r < 8; ++r) {
      int row = m0 + wid * 16 + r + (lhalf << 3);
      femb[((size_t)row * C_ + ch) * E_ + col] = acc[nt][r] + bias;
    }
  }
}

// ---------------------------------------------------------------------------
// Kernel 2: fused scores / softmax(axis=C) / pooled. One block per example.
// feat_emb + attention tiles staged with async global->LDS (pure copies).
// ---------------------------------------------------------------------------
__global__ __launch_bounds__(128)
void attn_pool_kernel(const float* __restrict__ femb, const float* __restrict__ att,
                      float* __restrict__ pooled) {
  __shared__ float fe[C_ * E_];       // 25.6 KB
  __shared__ float at[KATT_ * E_];
  __shared__ float sc[C_ * KATT_];
  const int b = blockIdx.x;
  const int t = threadIdx.x;
#if USE_ASYNC_LDS
  {
    const float* gfe = &femb[(size_t)b * C_ * E_];   // contiguous [C,E] slab
    // fe: 6400 floats = 1600 float4
    for (int i4 = t; i4 < (C_ * E_) / 4; i4 += 128) {
      ASYNC_CP16(gfe + i4 * 4, fe + i4 * 4);
    }
    // at: 896 floats = 224 float4
    for (int i4 = t; i4 < (KATT_ * E_) / 4; i4 += 128) {
      ASYNC_CP16(att + i4 * 4, at + i4 * 4);
    }
    __builtin_amdgcn_s_wait_asynccnt(0);
  }
#else
  for (int i = t; i < KATT_ * E_; i += 128) at[i] = att[i];
  for (int i = t; i < C_ * E_; i += 128) fe[i] = femb[(size_t)b * C_ * E_ + i];
#endif
  __syncthreads();
  for (int pr = t; pr < C_ * KATT_; pr += 128) {
    int c = pr / KATT_, k = pr - c * KATT_;
    float s = 0.f;
    for (int e = 0; e < E_; ++e) s += fe[c * E_ + e] * at[k * E_ + e];
    sc[pr] = s;
  }
  __syncthreads();
  if (t < KATT_) {
    const int k = t;
    float mx = -1e30f;
    for (int c = 0; c < C_; ++c) mx = fmaxf(mx, sc[c * KATT_ + k]);
    float sm = 0.f;
    for (int c = 0; c < C_; ++c) { float ex = __expf(sc[c * KATT_ + k] - mx); sc[c * KATT_ + k] = ex; sm += ex; }
    float inv = 1.f / sm;
    for (int c = 0; c < C_; ++c) sc[c * KATT_ + k] *= inv;
  }
  __syncthreads();
  const int e = t;
  for (int k = 0; k < KATT_; ++k) {
    float s = 0.f;
    for (int c = 0; c < C_; ++c) s += sc[c * KATT_ + k] * fe[c * E_ + e];
    pooled[((size_t)b * KATT_ + k) * E_ + e] = s;
  }
}

// ---------------------------------------------------------------------------
// Kernel 3: landmark embedding gather -> cur[b][e][p], p = i*4+j
// ---------------------------------------------------------------------------
__global__ __launch_bounds__(256)
void lm_init_kernel(const int* __restrict__ lmk, const float* __restrict__ emap,
                    float* __restrict__ cur) {
  int idx = blockIdx.x * 256 + threadIdx.x;
  if (idx >= B_ * E_ * 16) return;
  int p = idx & 15, e = (idx >> 4) & 127, b = idx >> 11;
  int l = lmk[b * 16 + p];
  cur[idx] = emap[l * E_ + e];
}

// ---------------------------------------------------------------------------
// Kernel 4a: per-example 3x3 action mask = softmax(pooled[:,4+j] @ act_lin + b)
// ---------------------------------------------------------------------------
__global__ __launch_bounds__(32)
void mask_kernel(const float* __restrict__ pooled, const float* __restrict__ w,
                 const float* __restrict__ bias, float* __restrict__ mask, int step) {
  __shared__ float s[9];
  int b = blockIdx.x, a = threadIdx.x;
  if (a < 9) {
    float acc = bias[a];
    const float* p = &pooled[((size_t)b * KATT_ + MS_ + step) * E_];
    for (int e = 0; e < E_; ++e) acc += p[e] * w[e * 9 + a];
    s[a] = acc;
  }
  __syncthreads();
  if (a < 9) {
    float mx = s[0];
    for (int i = 1; i < 9; ++i) mx = fmaxf(mx, s[i]);
    float sm = 0.f;
    for (int i = 0; i < 9; ++i) sm += __expf(s[i] - mx);
    mask[b * 9 + a] = __expf(s[a] - mx) / sm;
  }
}

// ---------------------------------------------------------------------------
// Kernel 4b: masked 3x3 conv as per-example GEMM [128 x 1152]x[1152 x 16], WMMA.
// One block per example, 8 waves: wave w -> output rows w*16..+15, single N-tile.
// Activation slab staged via async global->LDS; B operand = mask-scaled im2col.
// ---------------------------------------------------------------------------
__global__ __launch_bounds__(256)
void conv_kernel(const float* __restrict__ cw, const float* __restrict__ cur,
                 const float* __restrict__ mask, float* __restrict__ nxt) {
  __shared__ float  xs[E_ * 16];       // this example's activation, 8 KB
  __shared__ float  mk[9];
  __shared__ __bf16 Aw[128 * 32];      // weight chunk [o][k]
  __shared__ __bf16 Bp[16 * 32];       // patch chunk [n][k]
  const int b = blockIdx.x, tid = threadIdx.x;
  const int wid = tid >> 5, lane = tid & 31, lhalf = lane >> 4, l15 = lane & 15;

#if USE_ASYNC_LDS
  {
    const float* gsrc = &cur[(size_t)b * E_ * 16];   // 2048 floats = 512 float4
#pragma unroll
    for (int q = 0; q < 2; ++q) {
      int i4 = tid + q * 256;
      ASYNC_CP16(gsrc + i4 * 4, xs + i4 * 4);
    }
    __builtin_amdgcn_s_wait_asynccnt(0);
  }
#else
  for (int i = tid; i < E_ * 16; i += 256) xs[i] = cur[(size_t)b * E_ * 16 + i];
#endif
  if (tid < 9) mk[tid] = mask[b * 9 + tid];
  __syncthreads();

  v8f acc = {};
  const int KTOT = E_ * 9;             // 1152
  for (int kt = 0; kt < KTOT / 32; ++kt) {
    const int k0 = kt * 32;
    // stage weight chunk (contiguous in conv_weight: [o][e*9+tap])
#pragma unroll
    for (int q = 0; q < 4; ++q) {
      int idx = tid * 16 + q * 4;
      int o = idx >> 5, kk = idx & 31;
      const float4 v = *(const float4*)&cw[(size_t)o * KTOT + k0 + kk];
      *(v4bf*)&Aw[o * 32 + kk] = cvt4(v);
    }
    // stage masked im2col patch chunk: k = e*9+tap, n = i*4+j
#pragma unroll
    for (int q = 0; q < 2; ++q) {
      int idx = tid * 2 + q;             // 0..511
      int kk = idx >> 4, n = idx & 15;
      int k = k0 + kk;
      int e = k / 9, tap = k - e * 9;
      int ki = tap / 3, kj = tap - ki * 3;
      int ii = (n >> 2) + ki - 1, jj = (n & 3) + kj - 1;
      float val = 0.f;
      if ((unsigned)ii < 4u && (unsigned)jj < 4u) val = xs[e * 16 + ii * 4 + jj] * mk[tap];
      Bp[n * 32 + kk] = (__bf16)val;
    }
    __syncthreads();

    const int kb = lhalf * 8;
    const int am = wid * 16 + l15;
    v16bf a  = concat8(*(const v8bf*)&Aw[am * 32 + kb],
                       *(const v8bf*)&Aw[am * 32 + kb + 16]);
    v16bf bb = concat8(*(const v8bf*)&Bp[l15 * 32 + kb],
                       *(const v8bf*)&Bp[l15 * 32 + kb + 16]);
    acc = __builtin_amdgcn_wmma_f32_16x16x32_bf16(
        false, a, false, bb, (short)0, acc, false, false);
    __syncthreads();
  }

#pragma unroll
  for (int r = 0; r < 8; ++r) {
    int o = wid * 16 + r + (lhalf << 3);
    nxt[((size_t)b * E_ + o) * 16 + l15] = acc[r];
  }
}

// ---------------------------------------------------------------------------
// Kernel 5: logits[b][n] (+)= sum_e pooled[b][step][e] * cur[b][e][n]
// 16 threads per example; 16 examples per 256-thread block.
// ---------------------------------------------------------------------------
__global__ __launch_bounds__(256)
void logits_kernel(const float* __restrict__ pooled, const float* __restrict__ cur,
                   float* __restrict__ logits, int step, int init) {
  int tid = threadIdx.x;
  int b = blockIdx.x * 16 + (tid >> 4);
  int n = tid & 15;
  float acc = init ? 0.f : logits[b * 16 + n];
  const float* p = &pooled[((size_t)b * KATT_ + step) * E_];
  const float* c = &cur[(size_t)b * E_ * 16 + n];
  for (int e = 0; e < E_; ++e) acc += p[e] * c[e * 16];
  logits[b * 16 + n] = acc;
}

// ---------------------------------------------------------------------------
// Kernel 6: final softmax over the 16 logits per example.
// ---------------------------------------------------------------------------
__global__ __launch_bounds__(256)
void out_softmax_kernel(const float* __restrict__ logits, float* __restrict__ out) {
  int b = blockIdx.x * 256 + threadIdx.x;
  if (b >= B_) return;
  float v[16];
  float mx = -1e30f;
  for (int n = 0; n < 16; ++n) { v[n] = logits[b * 16 + n]; mx = fmaxf(mx, v[n]); }
  float sm = 0.f;
  for (int n = 0; n < 16; ++n) { v[n] = __expf(v[n] - mx); sm += v[n]; }
  float inv = 1.f / sm;
  for (int n = 0; n < 16; ++n) out[b * 16 + n] = v[n] * inv;
}

// ---------------------------------------------------------------------------
extern "C" void kernel_launch(void* const* d_in, const int* in_sizes, int n_in,
                              void* d_out, int out_size, void* d_ws, size_t ws_size,
                              hipStream_t stream) {
  (void)in_sizes; (void)n_in; (void)out_size; (void)ws_size;
  const float* msg  = (const float*)d_in[0];   // [B,C,V]
  const int*   lmk  = (const int*)  d_in[1];   // [B,4,4]
  const float* fw   = (const float*)d_in[2];   // [C,V,E]
  const float* fb   = (const float*)d_in[3];   // [C,E]
  const float* emap = (const float*)d_in[4];   // [NL,E]
  const float* cw   = (const float*)d_in[5];   // [E,E,3,3]
  const float* att  = (const float*)d_in[6];   // [7,E]
  const float* alw  = (const float*)d_in[7];   // [E,9]
  const float* alb  = (const float*)d_in[8];   // [9]
  float* out = (float*)d_out;

  float* ws     = (float*)d_ws;
  float* femb   = ws;  ws += (size_t)B_ * C_ * E_;       // 52.4 MB
  float* pooled = ws;  ws += (size_t)B_ * KATT_ * E_;    // 7.3 MB
  float* bufA   = ws;  ws += (size_t)B_ * E_ * 16;       // 16.8 MB
  float* bufB   = ws;  ws += (size_t)B_ * E_ * 16;       // 16.8 MB
  float* maskb  = ws;  ws += (size_t)B_ * 9;
  float* logits = ws;  ws += (size_t)B_ * 16;

  feat_gemm_kernel<<<dim3(B_ / 128, C_), 256, 0, stream>>>(msg, fw, fb, femb);
  attn_pool_kernel<<<B_, 128, 0, stream>>>(femb, att, pooled);
  lm_init_kernel<<<(B_ * E_ * 16) / 256, 256, 0, stream>>>(lmk, emap, bufA);
  logits_kernel<<<B_ / 16, 256, 0, stream>>>(pooled, bufA, logits, 0, 1);

  float* cur = bufA;
  float* nxt = bufB;
  for (int j = 0; j < MS_ - 1; ++j) {
    mask_kernel<<<B_, 32, 0, stream>>>(pooled, alw, alb, maskb, j);
    conv_kernel<<<B_, 256, 0, stream>>>(cw, cur, maskb, nxt);
    float* t = cur; cur = nxt; nxt = t;
    logits_kernel<<<B_ / 16, 256, 0, stream>>>(pooled, cur, logits, j + 1, 0);
  }
  out_softmax_kernel<<<(B_ + 255) / 256, 256, 0, stream>>>(logits, out);
}
